// StochPooledConvolutionalNetwork_19370302505155
// MI455X (gfx1250) — matrix operand order repaired and checked by reading, not compiled, not run on hardware
//
#include <hip/hip_runtime.h>
#include <hip/hip_bf16.h>

// ---------------------------------------------------------------------------
// Types for CDNA5 WMMA (wave32): v_wmma_f32_16x16x32_bf16
// ---------------------------------------------------------------------------
typedef __bf16        v16bf __attribute__((ext_vector_type(16)));
typedef float         v8f   __attribute__((ext_vector_type(8)));

// ---------------------------------------------------------------------------
// Generic WMMA GEMM:  C[M,N] = op(A) * B (+ bias),  fp32 in/out, bf16 WMMA.
//   TA=false: A is [M,K] row-major.   TA=true: A is [K,M] row-major (A^T used).
// One wave computes one 16x16 C tile; 8 waves / block. Batched via blockIdx.y.
// CDNA5 operand layouts per ISA 7.12.2 (16-bit A 16x32, B 32x16, f32 C/D).
// Conversions rely on native v_cvt_*bf16* via __bf16 casts (1 VALU / pair).
// ---------------------------------------------------------------------------
template<bool TA>
__global__ __launch_bounds__(256)
void wmma_gemm_kernel(const float* __restrict__ A, const float* __restrict__ Bm,
                      const float* __restrict__ bias, float* __restrict__ C,
                      int M, int N, int K, long sA, long sB, long sC)
{
    const float* Ab = A  + (long)blockIdx.y * sA;
    const float* Bb = Bm + (long)blockIdx.y * sB;
    float*       Cb = C  + (long)blockIdx.y * sC;

    const int lane = threadIdx.x & 31;
    const int wv   = threadIdx.x >> 5;
    const int half = lane >> 4;     // 0: lanes 0-15, 1: lanes 16-31
    const int l15  = lane & 15;

    const int tilesN = N >> 4;
    const int tilesM = M >> 4;
    const int tile   = blockIdx.x * (blockDim.x >> 5) + wv;   // wave-uniform
    if (tile >= tilesM * tilesN) return;                       // whole-wave exit
    const int tm = tile / tilesN, tn = tile % tilesN;
    const int row0 = tm << 4, col0 = tn << 4;

    const int m = row0 + l15;   // A-matrix row owned by this lane
    const int n = col0 + l15;   // B/C/D column owned by this lane

    v8f acc = {};
    for (int k0 = 0; k0 < K; k0 += 32) {
        v16bf av, bv;
#pragma unroll
        for (int i = 0; i < 8; ++i) {
            // A 16x32 bf16 layout: V0-3 -> K = half*8 + {0..7}; V4-7 -> +16
            const int kA = k0 + ((i & 4) << 2) + (half << 3) + ((i & 3) << 1);
            float a0, a1;
            if (TA) {
                a0 = Ab[(long)kA * M + m];
                a1 = Ab[(long)(kA + 1) * M + m];
            } else {
                const float2 ap = *(const float2*)(Ab + (long)m * K + kA);
                a0 = ap.x; a1 = ap.y;
            }
            av[2 * i]     = (__bf16)a0;
            av[2 * i + 1] = (__bf16)a1;
            // B 32x16 bf16 layout: V0-3 -> K = half*8 + {0..7}; V4-7 -> +16
            const int kB = k0 + ((i & 4) << 2) + (half << 3) + ((i & 3) << 1);
            bv[2 * i]     = (__bf16)Bb[(long)kB * N + n];
            bv[2 * i + 1] = (__bf16)Bb[(long)(kB + 1) * N + n];
        }
        acc = __builtin_amdgcn_wmma_f32_16x16x32_bf16(
                  false, av, false, bv, (short)0, acc, false, false);
    }
    const float bb = bias ? bias[n] : 0.0f;
#pragma unroll
    for (int r = 0; r < 8; ++r) {   // D f32: VGPR r -> M = r + 8*half, N = lane&15
        Cb[(long)(row0 + r + (half << 3)) * N + n] = acc[r] + bb;
    }
}

// ---------------------------------------------------------------------------
// Elementwise / graph kernels
// ---------------------------------------------------------------------------
__global__ void zero_kernel(float* p, long n) {
    long i = (long)blockIdx.x * blockDim.x + threadIdx.x;
    if (i < n) p[i] = 0.0f;
}

__global__ void deg_kernel(const int* __restrict__ dst, float* __restrict__ deg, int E) {
    int e = blockIdx.x * blockDim.x + threadIdx.x;
    if (e < E) atomicAdd(&deg[dst[e]], 1.0f);
}

__global__ void dinv_kernel(const float* __restrict__ deg, float* __restrict__ dinv, int n) {
    int i = blockIdx.x * blockDim.x + threadIdx.x;
    if (i < n) { float d = deg[i] + 1.0f; dinv[i] = d > 0.0f ? rsqrtf(d) : 0.0f; }
}

// One wave per edge; lane handles 4 of 128 channels (float4 gather, 4 atomics).
__global__ __launch_bounds__(256)
void edge_agg_kernel(const float* __restrict__ P, const int* __restrict__ src,
                     const int* __restrict__ dst, const float* __restrict__ dinv,
                     float* __restrict__ Q, int E, int useCoef)
{
    int wave = (int)(((long)blockIdx.x * blockDim.x + threadIdx.x) >> 5);
    int lane = threadIdx.x & 31;
    if (wave >= E) return;
    int s = src[wave], d = dst[wave];
    float coef = useCoef ? dinv[s] * dinv[d] : 1.0f;
    const float4 v = *(const float4*)(P + (long)s * 128 + lane * 4);
    float* q = Q + (long)d * 128 + lane * 4;
    atomicAdd(q + 0, v.x * coef);
    atomicAdd(q + 1, v.y * coef);
    atomicAdd(q + 2, v.z * coef);
    atomicAdd(q + 3, v.w * coef);
}

// Q += P * dinv^2 (self loop) + bias   over [N,128]
__global__ void self_bias_kernel(float* __restrict__ Q, const float* __restrict__ P,
                                 const float* __restrict__ dinv,
                                 const float* __restrict__ bias, long NC)
{
    long i = (long)blockIdx.x * blockDim.x + threadIdx.x;
    if (i >= NC) return;
    int node = (int)(i >> 7), c = (int)(i & 127);
    float di = dinv[node];
    Q[i] += P[i] * di * di + bias[c];
}

// Per-channel sum / sum-of-squares over R rows of [R,128] -> stats[0:128], stats[128:256]
__global__ __launch_bounds__(128)
void bn_stats_kernel(const float* __restrict__ X, float* __restrict__ stats, long R)
{
    int c = threadIdx.x;              // blockDim.x == 128
    float s = 0.0f, s2 = 0.0f;
    for (long r = blockIdx.x; r < R; r += gridDim.x) {
        float v = X[r * 128 + c]; s += v; s2 += v * v;
    }
    atomicAdd(&stats[c], s);
    atomicAdd(&stats[128 + c], s2);
}

// out = (res? res : 0) + silu(g*(x-m)*rsqrt(v+eps)+be)
__global__ void bn_apply_kernel(const float* __restrict__ X, const float* __restrict__ res,
                                const float* __restrict__ stats, const float* __restrict__ g,
                                const float* __restrict__ be, float* __restrict__ out, long R)
{
    long i = (long)blockIdx.x * blockDim.x + threadIdx.x;
    if (i >= R * 128) return;
    int c = (int)(i & 127);
    float m  = stats[c] / (float)R;
    float vv = stats[128 + c] / (float)R - m * m;
    float xn = g[c] * (X[i] - m) * rsqrtf(vv + 1e-5f) + be[c];
    float sil = xn * (1.0f / (1.0f + __expf(-xn)));
    out[i] = (res ? res[i] : 0.0f) + sil;
}

// Row softmax, one wave per row, width W in {32,128}
__global__ __launch_bounds__(256)
void softmax_rows_kernel(const float* __restrict__ X, float* __restrict__ out, long R, int W)
{
    long wave = ((long)blockIdx.x * blockDim.x + threadIdx.x) >> 5;
    int lane = threadIdx.x & 31;
    if (wave >= R) return;
    const int vpt = W >> 5;           // 1 or 4
    const float* x = X + wave * W;
    float v[4]; float mx = -3.0e38f;
    for (int j = 0; j < vpt; ++j) { v[j] = x[lane * vpt + j]; mx = fmaxf(mx, v[j]); }
    for (int o = 16; o > 0; o >>= 1) mx = fmaxf(mx, __shfl_xor(mx, o, 32));
    float sum = 0.0f;
    for (int j = 0; j < vpt; ++j) { v[j] = __expf(v[j] - mx); sum += v[j]; }
    for (int o = 16; o > 0; o >>= 1) sum += __shfl_xor(sum, o, 32);
    float inv = 1.0f / sum;
    for (int j = 0; j < vpt; ++j) out[wave * W + lane * vpt + j] = v[j] * inv;
}

// Dense graphs: dinv2[row] = rsqrt(1 + rowsum(A2[row,:])) ; one wave per row of 128
__global__ __launch_bounds__(256)
void rowsum_dinv2_kernel(const float* __restrict__ A2, float* __restrict__ dinv2, long Rows)
{
    long wave = ((long)blockIdx.x * blockDim.x + threadIdx.x) >> 5;
    int lane = threadIdx.x & 31;
    if (wave >= Rows) return;
    float s = 0.0f;
    for (int j = 0; j < 4; ++j) s += A2[wave * 128 + lane + 32 * j];
    for (int o = 16; o > 0; o >>= 1) s += __shfl_xor(s, o, 32);
    float d = s + 1.0f;
    if (lane == 0) dinv2[wave] = d > 0.0f ? rsqrtf(d) : 0.0f;
}

// An[b,i,j] = dinv2[b,i] * (A2[b,i,j] + (i==j)) * dinv2[b,j]
__global__ void an_kernel(const float* __restrict__ A2, const float* __restrict__ dinv2,
                          float* __restrict__ An, long total)
{
    long i = (long)blockIdx.x * blockDim.x + threadIdx.x;
    if (i >= total) return;
    long b = i >> 14; int rest = (int)(i & 16383);
    int r = rest >> 7, c = rest & 127;
    float a = A2[i] + (r == c ? 1.0f : 0.0f);
    An[i] = dinv2[b * 128 + r] * a * dinv2[b * 128 + c];
}

// g[b,c] = mean over 32 rows of x3[b,:,c]
__global__ void mean_rows_kernel(const float* __restrict__ x3, float* __restrict__ g, int total)
{
    int i = blockIdx.x * blockDim.x + threadIdx.x;   // over B*128
    if (i >= total) return;
    int b = i >> 7, c = i & 127;
    float s = 0.0f;
    for (int r = 0; r < 32; ++r) s += x3[((long)b * 32 + r) * 128 + c];
    g[i] = s * (1.0f / 32.0f);
}

// Final: out = log_softmax(g @ wl + bl); also writes the aux zero scalar.
__global__ void head_kernel(const float* __restrict__ g, const float* __restrict__ wl,
                            const float* __restrict__ bl, float* __restrict__ out, int B)
{
    int b = threadIdx.x;
    if (b < B) {
        float z[10];
        for (int o = 0; o < 10; ++o) {
            float a = bl[o];
            for (int c = 0; c < 128; ++c) a += g[b * 128 + c] * wl[c * 10 + o];
            z[o] = a;
        }
        float mx = z[0];
        for (int o = 1; o < 10; ++o) mx = fmaxf(mx, z[o]);
        float s = 0.0f;
        for (int o = 0; o < 10; ++o) s += __expf(z[o] - mx);
        float lse = logf(s) + mx;
        for (int o = 0; o < 10; ++o) out[b * 10 + o] = z[o] - lse;
    }
    if (threadIdx.x == 0) out[B * 10] = 0.0f;   // aux loss is exact zero
}

// ---------------------------------------------------------------------------
// Host orchestration
// ---------------------------------------------------------------------------
extern "C" void kernel_launch(void* const* d_in, const int* in_sizes, int n_in,
                              void* d_out, int out_size, void* d_ws, size_t ws_size,
                              hipStream_t stream)
{
    const float* x   = (const float*)d_in[0];
    const int*   ei  = (const int*)  d_in[1];
    const float* w1a = (const float*)d_in[4],  *b1a = (const float*)d_in[5];
    const float* g1a = (const float*)d_in[6],  *be1a= (const float*)d_in[7];
    const float* w1b = (const float*)d_in[8],  *b1b = (const float*)d_in[9];
    const float* g1b = (const float*)d_in[10], *be1b= (const float*)d_in[11];
    const float* wp1 = (const float*)d_in[12], *bp1 = (const float*)d_in[13];
    const float* w2a = (const float*)d_in[14], *b2a = (const float*)d_in[15];
    const float* g2a = (const float*)d_in[16], *be2a= (const float*)d_in[17];
    const float* w2b = (const float*)d_in[18], *b2b = (const float*)d_in[19];
    const float* g2b = (const float*)d_in[20], *be2b= (const float*)d_in[21];
    const float* wp2 = (const float*)d_in[22], *bp2 = (const float*)d_in[23];
    const float* wl  = (const float*)d_in[24], *bl  = (const float*)d_in[25];
    float* out = (float*)d_out;

    const int CIN = 64, C = 128, K1 = 128, K2 = 32;
    const int N   = in_sizes[0] / CIN;       // 32768
    const int E   = in_sizes[1] / 2;         // 524288
    const int B   = in_sizes[3] - 1;         // 32
    const int NPG = N / B;                   // 1024
    const long NC = (long)N * C;
    const int* esrc = ei;
    const int* edst = ei + E;

    // workspace layout (floats)
    float* w = (float*)d_ws;
    size_t off = 0;
    auto alloc = [&](size_t n) { float* p = w + off; off += (n + 255) & ~(size_t)255; return p; };
    float* deg   = alloc(N);
    float* dinv  = alloc(N);
    float* P     = alloc(NC);            // GEMM outputs (pre-aggregation)
    float* Q     = alloc(NC);            // aggregation outputs
    float* H     = alloc(NC);            // activations
    float* S     = alloc(NC);            // pooling assignment softmax
    float* T     = alloc(NC);            // t = A s
    float* x2    = alloc((size_t)B * K1 * C);
    float* A2    = alloc((size_t)B * K1 * K1);
    float* An    = alloc((size_t)B * K1 * K1);
    float* hw    = alloc((size_t)B * K1 * C);
    float* g2b_  = alloc((size_t)B * K1 * C);
    float* h2    = alloc((size_t)B * K1 * C);
    float* s2    = alloc((size_t)B * K1 * K2);
    float* x3    = alloc((size_t)B * K2 * C);
    float* gm    = alloc((size_t)B * C);
    float* stats = alloc(256);
    float* dinv2 = alloc((size_t)B * K1);
    (void)ws_size; (void)n_in; (void)out_size;

    auto zero = [&](float* p, long n) {
        zero_kernel<<<(int)((n + 255) / 256), 256, 0, stream>>>(p, n);
    };
    auto gemm = [&](bool ta, const float* A, const float* Bm, const float* bias, float* Cp,
                    int M, int Nc, int K, long sA, long sB, long sC, int batch) {
        int tiles = (M / 16) * (Nc / 16);
        dim3 grid((tiles + 7) / 8, batch);
        if (ta) wmma_gemm_kernel<true ><<<grid, 256, 0, stream>>>(A, Bm, bias, Cp, M, Nc, K, sA, sB, sC);
        else    wmma_gemm_kernel<false><<<grid, 256, 0, stream>>>(A, Bm, bias, Cp, M, Nc, K, sA, sB, sC);
    };
    auto agg = [&](const float* src_feat, float* dst_acc, int useCoef) {
        zero(dst_acc, NC);
        edge_agg_kernel<<<(E * 32 + 255) / 256, 256, 0, stream>>>(src_feat, esrc, edst, dinv, dst_acc, E, useCoef);
    };
    auto bn = [&](const float* X, const float* res, const float* gg, const float* bb,
                  float* o, long R) {
        zero(stats, 256);
        int nb = R >= 4096 ? 1024 : 256;
        bn_stats_kernel<<<nb, 128, 0, stream>>>(X, stats, R);
        bn_apply_kernel<<<(int)((R * 128 + 255) / 256), 256, 0, stream>>>(X, res, stats, gg, bb, o, R);
    };

    // ---- degrees / norm coefficients ----
    zero(deg, N);
    deg_kernel<<<(E + 255) / 256, 256, 0, stream>>>(edst, deg, E);
    dinv_kernel<<<(N + 255) / 256, 256, 0, stream>>>(deg, dinv, N);

    // ---- sparse stack ----
    gemm(false, x, w1a, nullptr, P, N, C, CIN, 0, 0, 0, 1);
    agg(P, Q, 1);
    self_bias_kernel<<<(int)((NC + 255) / 256), 256, 0, stream>>>(Q, P, dinv, b1a, NC);
    bn(Q, nullptr, g1a, be1a, H, N);

    gemm(false, H, w1b, nullptr, P, N, C, C, 0, 0, 0, 1);
    agg(P, Q, 1);
    self_bias_kernel<<<(int)((NC + 255) / 256), 256, 0, stream>>>(Q, P, dinv, b1b, NC);
    bn(Q, H, g1b, be1b, H, N);                       // residual add, in-place OK

    gemm(false, H, wp1, nullptr, P, N, K1, C, 0, 0, 0, 1);
    agg(P, Q, 1);
    self_bias_kernel<<<(int)((NC + 255) / 256), 256, 0, stream>>>(Q, P, dinv, bp1, NC);
    softmax_rows_kernel<<<(int)((N + 7) / 8), 256, 0, stream>>>(Q, S, N, K1);

    agg(S, T, 0);                                    // t = A s (no norm, no self)

    // ---- pooling: x2 = S^T H, A2 = S^T T per graph ----
    gemm(true, S, H, nullptr, x2, K1, C, NPG, (long)NPG * K1, (long)NPG * C, (long)K1 * C, B);
    gemm(true, S, T, nullptr, A2, K1, K1, NPG, (long)NPG * K1, (long)NPG * K1, (long)K1 * K1, B);

    // ---- dense adjacency normalization ----
    rowsum_dinv2_kernel<<<(B * K1 * 32 + 255) / 256, 256, 0, stream>>>(A2, dinv2, (long)B * K1);
    an_kernel<<<(int)(((long)B * K1 * K1 + 255) / 256), 256, 0, stream>>>(A2, dinv2, An, (long)B * K1 * K1);

    // ---- dense stack ----
    gemm(false, x2, w2a, nullptr, hw, B * K1, C, C, 0, 0, 0, 1);
    gemm(false, An, hw, b2a, g2b_, K1, C, K1, (long)K1 * K1, (long)K1 * C, (long)K1 * C, B);
    bn(g2b_, nullptr, g2a, be2a, h2, (long)B * K1);

    gemm(false, h2, w2b, nullptr, hw, B * K1, C, C, 0, 0, 0, 1);
    gemm(false, An, hw, b2b, g2b_, K1, C, K1, (long)K1 * K1, (long)K1 * C, (long)K1 * C, B);
    bn(g2b_, h2, g2b, be2b, h2, (long)B * K1);

    gemm(false, h2, wp2, nullptr, hw, B * K1, K2, C, 0, 0, 0, 1);   // [B*K1,32]
    gemm(false, An, hw, bp2, s2, K1, K2, K1, (long)K1 * K1, (long)K1 * K2, (long)K1 * K2, B);
    softmax_rows_kernel<<<(B * K1 + 7) / 8, 256, 0, stream>>>(s2, s2, (long)B * K1, K2);

    // ---- x3 = s2^T h2 per graph -> [B,32,128]; mean pool; head ----
    gemm(true, s2, h2, nullptr, x3, K2, C, K1, (long)K1 * K2, (long)K1 * C, (long)K2 * C, B);
    mean_rows_kernel<<<(B * C + 255) / 256, 256, 0, stream>>>(x3, gm, B * C);
    head_kernel<<<1, 64, 0, stream>>>(gm, wl, bl, out, B);
}
